// MemoryEfficientMultiHeadAttention_61306363183427
// MI455X (gfx1250) — matrix operand order repaired and checked by reading, not compiled, not run on hardware
//
#include <hip/hip_runtime.h>
#include <hip/hip_bf16.h>
#include <stdint.h>

typedef __bf16 bf16_t;
typedef __attribute__((ext_vector_type(16))) __bf16 bf16x16;
typedef __attribute__((ext_vector_type(8)))  __bf16 bf16x8;
typedef __attribute__((ext_vector_type(8)))  float  f32x8;
typedef __attribute__((ext_vector_type(4)))  float  f32x4;
typedef uint32_t u32;
typedef __attribute__((ext_vector_type(4))) u32 u32x4;
typedef __attribute__((ext_vector_type(8))) u32 u32x8;

#define D_MODEL 1024
#define SEQ     2048
#define BATCH   4
#define HEADS   16
#define DEPTH   64
#define M_TOTAL (BATCH * SEQ)   // 8192

// LDS row strides (bf16 elements), padded to dodge bank conflicts on b128 gathers
#define LDA 40   // A tile rows: 32 k + 8 pad
#define LDB 40   // B tile rows: 32 k + 8 pad
#define LDK 72   // K tile rows: 64 d + 8 pad   (TDM pad: 32 DW interval, 4 DW pad)
#define LDV 40   // Vt tile rows: 32 s + 8 pad  (TDM pad: 16 DW interval, 4 DW pad)

static __device__ __forceinline__ bf16x16 cat8(bf16x8 lo, bf16x8 hi) {
  return __builtin_shufflevector(lo, hi, 0,1,2,3,4,5,6,7,8,9,10,11,12,13,14,15);
}

static __device__ __forceinline__ bf16x8 cvt8(f32x4 a, f32x4 b) {
  bf16x8 r;
  r[0]=(bf16_t)a[0]; r[1]=(bf16_t)a[1]; r[2]=(bf16_t)a[2]; r[3]=(bf16_t)a[3];
  r[4]=(bf16_t)b[0]; r[5]=(bf16_t)b[1]; r[6]=(bf16_t)b[2]; r[7]=(bf16_t)b[3];
  return r;
}

// --------------------------------------------------------------------------
// TDM: async 2D tile DMA (global -> LDS) with LDS padding done by the mover.
// D# fields per CDNA5 ISA §8.3/§8.4. 2-SGPR-group form (2D tensor). Issued
// wave-uniform; tracked with TENSORcnt.
// --------------------------------------------------------------------------
static __device__ __forceinline__ void tdm_load_2d(
    u32 ldsOff, const void* gptr, u32 tileW, u32 tileH,
    u32 strideElems, u32 padIntCode, u32 padAmtCode) {
  const uint64_t ga = (uint64_t)(uintptr_t)gptr;
  u32x4 g0;
  g0[0] = 1u;                              // count=1 (valid), user mode
  g0[1] = ldsOff;                          // LDS byte address
  g0[2] = (u32)ga;                         // global_addr[31:0]
  g0[3] = (u32)(ga >> 32) | (2u << 30);    // global_addr[56:32] | type=2
  u32x8 g1;
  g1[0] = (1u << 16)                       // data_size = 2 bytes (bf16)
        | (1u << 20)                       // pad_enable
        | (padIntCode << 22) | (padAmtCode << 25);
  g1[1] = (strideElems & 0xFFFFu) << 16;               // tensor_dim0 lo16
  g1[2] = ((strideElems >> 16) & 0xFFFFu)              // tensor_dim0 hi16
        | (0xFFFFu << 16);                             // tensor_dim1 lo16 (huge)
  g1[3] = 0xFu | (tileW << 16);                        // tensor_dim1 hi | tile_dim0
  g1[4] = tileH;                                       // tile_dim1 (tile_dim2 = 0)
  g1[5] = strideElems;                                 // tensor_dim0_stride lo32
  g1[6] = 0;                                           // stride hi | dim1_stride lo
  g1[7] = 0;
  asm volatile("tensor_load_to_lds %0, %1" :: "s"(g0), "s"(g1) : "memory");
}

// One K=32 step: load 2 A-frags + 2 B-frags from LDS, 4 WMMAs into acc[2][2].
static __device__ __forceinline__ void tile_mma(const bf16_t* __restrict__ Alds,
                                                const bf16_t* __restrict__ Blds,
                                                int waveM, int waveN, int l15, int half,
                                                f32x8 acc[2][2]) {
  bf16x16 af[2], bfg[2];
#pragma unroll
  for (int si = 0; si < 2; ++si) {
    // A 16x32: lane=row; lane<16: k {0..7,16..23}; lane>=16: k {8..15,24..31}
    const bf16_t* p = Alds + (waveM*32 + si*16 + l15)*LDA + half*8;
    af[si] = cat8(*(const bf16x8*)p, *(const bf16x8*)(p + 16));
  }
#pragma unroll
  for (int sj = 0; sj < 2; ++sj) {
    // B 32x16: lane=col; lane<16: k {0..7,16..23}; lane>=16: k {8..15,24..31}
    const bf16_t* p = Blds + (waveN*32 + sj*16 + l15)*LDB + half*8;
    bfg[sj] = cat8(*(const bf16x8*)p, *(const bf16x8*)(p + 16));
  }
#pragma unroll
  for (int si = 0; si < 2; ++si)
#pragma unroll
    for (int sj = 0; sj < 2; ++sj)
      acc[si][sj] = __builtin_amdgcn_wmma_f32_16x16x32_bf16(
          false, af[si], false, bfg[sj], (short)0, acc[si][sj], false, false);
}

// ---------------------------------------------------------------------------
// Kernel 1: fused Q/K/V projections.  y = x @ W^T + b  (bf16 WMMA, f32 accum)
// Software-pipelined: global prefetch of tile k+1 overlaps WMMAs of tile k,
// double-buffered LDS -> one barrier per K-step.
// grid = (N/64, M/128, 3); block = 256 (8 waves, 4x2 wave grid, 32x32/wave)
// z=0 -> Q (scaled 1/8, [b][h][s][d]); z=1 -> K; z=2 -> V^T [b][h][d][s].
// ---------------------------------------------------------------------------
__global__ __launch_bounds__(256) void gemm_qkv_kernel(
    const float* __restrict__ x,
    const float* __restrict__ wq, const float* __restrict__ bq,
    const float* __restrict__ wk, const float* __restrict__ bk,
    const float* __restrict__ wv, const float* __restrict__ bv,
    bf16_t* __restrict__ qws, bf16_t* __restrict__ kws, bf16_t* __restrict__ vtws) {
  __shared__ __align__(16) bf16_t Alds[2][128 * LDA];
  __shared__ __align__(16) bf16_t Blds[2][64 * LDB];

  const int z = blockIdx.z;
  const float* __restrict__ W  = (z == 0) ? wq : (z == 1) ? wk : wv;
  const float* __restrict__ Bb = (z == 0) ? bq : (z == 1) ? bk : bv;

  const int tid  = threadIdx.x;
  const int lane = tid & 31, wid = tid >> 5;
  const int waveM = wid >> 1, waveN = wid & 1;
  const int l15 = lane & 15, half = lane >> 4;
  const int mBase = blockIdx.y * 128, nBase = blockIdx.x * 64;

  f32x8 acc[2][2];
#pragma unroll
  for (int i = 0; i < 2; ++i)
#pragma unroll
    for (int j = 0; j < 2; ++j)
#pragma unroll
      for (int r = 0; r < 8; ++r) acc[i][j][r] = 0.f;

  const int rA = tid >> 1, kpA = (tid & 1) * 16;   // A stage: 16 floats/thread
  const int nB = tid >> 2, kpB = (tid & 3) * 8;    // B stage: 8 floats/thread
  const float* ap = x + (size_t)(mBase + rA) * D_MODEL + kpA;
  const float* bp = W + (size_t)(nBase + nB) * D_MODEL + kpB;

  f32x4 ar[4], br[2];
#pragma unroll
  for (int i = 0; i < 4; ++i) ar[i] = *(const f32x4*)(ap + i*4);
  br[0] = *(const f32x4*)bp;
  br[1] = *(const f32x4*)(bp + 4);

  for (int k0 = 0; k0 < D_MODEL; k0 += 32) {
    const int buf = (k0 >> 5) & 1;
    bf16_t* Al = &Alds[buf][0];
    bf16_t* Bl = &Blds[buf][0];
    *(bf16x8*)(Al + rA*LDA + kpA)     = cvt8(ar[0], ar[1]);
    *(bf16x8*)(Al + rA*LDA + kpA + 8) = cvt8(ar[2], ar[3]);
    *(bf16x8*)(Bl + nB*LDB + kpB)     = cvt8(br[0], br[1]);
    __syncthreads();
    if (k0 + 32 < D_MODEL) {     // prefetch next tile while WMMAs run
#pragma unroll
      for (int i = 0; i < 4; ++i) ar[i] = *(const f32x4*)(ap + k0 + 32 + i*4);
      br[0] = *(const f32x4*)(bp + k0 + 32);
      br[1] = *(const f32x4*)(bp + k0 + 36);
    }
    tile_mma(Al, Bl, waveM, waveN, l15, half, acc);
  }

  const float scale = (z == 0) ? 0.125f : 1.0f;   // fold 1/sqrt(depth) into Q
#pragma unroll
  for (int si = 0; si < 2; ++si)
#pragma unroll
    for (int sj = 0; sj < 2; ++sj) {
      const int col = nBase + waveN*32 + sj*16 + l15;
      const float bias = Bb[col];
      const int h = col >> 6, d = col & 63;
#pragma unroll
      for (int r = 0; r < 8; ++r) {
        const int row = mBase + waveM*32 + si*16 + half*8 + r;
        const int b = row >> 11, s = row & 2047;
        const float v = (acc[si][sj][r] + bias) * scale;
        if (z == 2) {
          vtws[((size_t)(b*HEADS + h)*DEPTH + d)*SEQ + s] = (bf16_t)v;
        } else {
          bf16_t* dst = (z == 0) ? qws : kws;
          dst[((size_t)(b*HEADS + h)*SEQ + s)*DEPTH + d] = (bf16_t)v;
        }
      }
    }
}

// ---------------------------------------------------------------------------
// Kernel 2: flash attention.  grid = (SEQ/128, B*H); block = 256 (8 waves).
// K/V^T tiles staged by the Tensor Data Mover (double-buffered, TENSORcnt),
// overlapping the DMA of block t+1 with the 8 WMMAs + softmax of block t.
// Computes S^T = K·Q^T so the score accumulator is already in WMMA-A layout
// for P·V (no cross-lane transpose).
// ---------------------------------------------------------------------------
__global__ __launch_bounds__(256) void attn_kernel(
    const bf16_t* __restrict__ Q, const bf16_t* __restrict__ K,
    const bf16_t* __restrict__ Vt, bf16_t* __restrict__ att) {
  __shared__ __align__(16) bf16_t Klds [2][32 * LDK];   // [key][d]
  __shared__ __align__(16) bf16_t Vtlds[2][64 * LDV];   // [d][key]

  const int tid  = threadIdx.x;
  const int lane = tid & 31, wid = tid >> 5;
  const int l15 = lane & 15, half = lane >> 4;
  const int bh = blockIdx.y;
  const int qBase = blockIdx.x * 128 + wid * 16;
  const size_t bhOff = (size_t)bh * SEQ * DEPTH;        // same stride for Q/K/Vt

  const bf16_t* Kb  = K  + bhOff;
  const bf16_t* Vtb = Vt + bhOff;

  auto issue_tile = [&](int kb, int buf) {
    // K tile 32 rows x 64 d, row stride 64; pad 4 DW every 32 DW -> LDK=72
    tdm_load_2d((u32)(uintptr_t)&Klds[buf][0], Kb + (size_t)kb * DEPTH,
                /*tileW*/DEPTH, /*tileH*/32, /*stride*/DEPTH, /*padInt*/4, /*padAmt*/3);
    // V^T tile 64 rows x 32 s, row stride SEQ; pad 4 DW every 16 DW -> LDV=40
    tdm_load_2d((u32)(uintptr_t)&Vtlds[buf][0], Vtb + kb,
                /*tileW*/32, /*tileH*/DEPTH, /*stride*/SEQ, /*padInt*/3, /*padAmt*/3);
  };

  if (wid == 0) issue_tile(0, 0);

  // Q B-frags (d 0..31, 32..63): b[e] = Q[q][d], lane=col(q), d {8h..8h+7, 16+8h..16+8h+7}.
  bf16x16 bqf[2];
  {
    const bf16_t* qp = Q + bhOff + (size_t)(qBase + l15)*DEPTH + half*8;
#pragma unroll
    for (int f = 0; f < 2; ++f)
      bqf[f] = cat8(*(const bf16x8*)(qp + f*32), *(const bf16x8*)(qp + f*32 + 16));
  }

  f32x8 o[4];
#pragma unroll
  for (int j = 0; j < 4; ++j)
#pragma unroll
    for (int r = 0; r < 8; ++r) o[j][r] = 0.f;
  float m_run = -1e30f, l_run = 0.f;

  for (int kb = 0; kb < SEQ; kb += 32) {
    const int buf = (kb >> 5) & 1;
    if (wid == 0) __builtin_amdgcn_s_wait_tensorcnt(0);  // tile kb landed
    __syncthreads();                                     // visible to all waves
    if (wid == 0 && kb + 32 < SEQ) issue_tile(kb + 32, buf ^ 1);  // overlap DMA

    const bf16_t* Kl  = &Klds[buf][0];
    const bf16_t* Vtl = &Vtlds[buf][0];

    // S^T frags: rows = keys (two 16-key subtiles), cols = queries
    f32x8 st[2];
#pragma unroll
    for (int s = 0; s < 2; ++s) {
      const bf16_t* kp_ = Kl + (s*16 + l15)*LDK + half*8;
      bf16x16 ka0 = cat8(*(const bf16x8*)kp_,        *(const bf16x8*)(kp_ + 16));
      bf16x16 ka1 = cat8(*(const bf16x8*)(kp_ + 32), *(const bf16x8*)(kp_ + 48));
      f32x8 zacc;
#pragma unroll
      for (int r = 0; r < 8; ++r) zacc[r] = 0.f;
      st[s] = __builtin_amdgcn_wmma_f32_16x16x32_bf16(false, ka0, false, bqf[0],
                                                      (short)0, zacc, false, false);
      st[s] = __builtin_amdgcn_wmma_f32_16x16x32_bf16(false, ka1, false, bqf[1],
                                                      (short)0, st[s], false, false);
    }

    // Online softmax per query (query = lane&15; stats replicated lanes q,q+16)
    float cmax = -1e30f;
#pragma unroll
    for (int s = 0; s < 2; ++s)
#pragma unroll
      for (int r = 0; r < 8; ++r) cmax = fmaxf(cmax, st[s][r]);
    cmax = fmaxf(cmax, __shfl_xor(cmax, 16, 32));
    const float mnew  = fmaxf(m_run, cmax);
    const float alpha = __expf(m_run - mnew);

    float lsum = 0.f;
    bf16x16 pa;
#pragma unroll
    for (int s = 0; s < 2; ++s)
#pragma unroll
      for (int r = 0; r < 8; ++r) {
        float p = __expf(st[s][r] - mnew);
        lsum += p;
        pa[s*8 + r] = (bf16_t)p;   // S^T C-layout == P A-layout: no shuffles
      }
    lsum += __shfl_xor(lsum, 16, 32);
    l_run = l_run * alpha + lsum;
    m_run = mnew;

    // Rescale O: O-frag row q = r + 8*half; alpha[q] lives at lane q
#pragma unroll
    for (int r = 0; r < 8; ++r) {
      float arw = __shfl(alpha, r + half*8, 32);
#pragma unroll
      for (int j = 0; j < 4; ++j) o[j][r] *= arw;
    }

    // O += P @ V  (V B-frags from transposed LDS tile: contiguous keys)
#pragma unroll
    for (int j = 0; j < 4; ++j) {
      const bf16_t* vp = Vtl + (j*16 + l15)*LDV + half*8;
      bf16x16 vb = cat8(*(const bf16x8*)vp, *(const bf16x8*)(vp + 16));
      o[j] = __builtin_amdgcn_wmma_f32_16x16x32_bf16(false, pa, false, vb,
                                                     (short)0, o[j], false, false);
    }
    __syncthreads();   // all reads of buf done before TDM (kb+64) overwrites it
  }

  // Normalize and store attended as [b][s][h*64+d] (bf16) for the dense GEMM
  const float linv = 1.0f / l_run;
  const int b = bh >> 4, h = bh & 15;
#pragma unroll
  for (int r = 0; r < 8; ++r) {
    const float lr = __shfl(linv, r + half*8, 32);
    const int q = qBase + half*8 + r;
#pragma unroll
    for (int j = 0; j < 4; ++j) {
      const int col = h*DEPTH + j*16 + l15;
      att[((size_t)b*SEQ + q)*D_MODEL + col] = (bf16_t)(o[j][r] * lr);
    }
  }
}

// ---------------------------------------------------------------------------
// Kernel 3: output projection.  out = attended @ dense_w^T + dense_b (fp32)
// Same pipelined double-buffered structure as kernel 1; A is already bf16.
// ---------------------------------------------------------------------------
__global__ __launch_bounds__(256) void gemm_dense_kernel(
    const bf16_t* __restrict__ A, const float* __restrict__ W,
    const float* __restrict__ bias, float* __restrict__ out) {
  __shared__ __align__(16) bf16_t Alds[2][128 * LDA];
  __shared__ __align__(16) bf16_t Blds[2][64 * LDB];

  const int tid  = threadIdx.x;
  const int lane = tid & 31, wid = tid >> 5;
  const int waveM = wid >> 1, waveN = wid & 1;
  const int l15 = lane & 15, half = lane >> 4;
  const int mBase = blockIdx.y * 128, nBase = blockIdx.x * 64;

  f32x8 acc[2][2];
#pragma unroll
  for (int i = 0; i < 2; ++i)
#pragma unroll
    for (int j = 0; j < 2; ++j)
#pragma unroll
      for (int r = 0; r < 8; ++r) acc[i][j][r] = 0.f;

  const int rA = tid >> 1, kpA = (tid & 1) * 16;
  const int nB = tid >> 2, kpB = (tid & 3) * 8;
  const bf16_t* ap = A + (size_t)(mBase + rA) * D_MODEL + kpA;
  const float*  bp = W + (size_t)(nBase + nB) * D_MODEL + kpB;

  bf16x8 arb[2];
  f32x4  br[2];
  arb[0] = *(const bf16x8*)ap;
  arb[1] = *(const bf16x8*)(ap + 8);
  br[0]  = *(const f32x4*)bp;
  br[1]  = *(const f32x4*)(bp + 4);

  for (int k0 = 0; k0 < D_MODEL; k0 += 32) {
    const int buf = (k0 >> 5) & 1;
    bf16_t* Al = &Alds[buf][0];
    bf16_t* Bl = &Blds[buf][0];
    *(bf16x8*)(Al + rA*LDA + kpA)     = arb[0];
    *(bf16x8*)(Al + rA*LDA + kpA + 8) = arb[1];
    *(bf16x8*)(Bl + nB*LDB + kpB)     = cvt8(br[0], br[1]);
    __syncthreads();
    if (k0 + 32 < D_MODEL) {
      arb[0] = *(const bf16x8*)(ap + k0 + 32);
      arb[1] = *(const bf16x8*)(ap + k0 + 40);
      br[0]  = *(const f32x4*)(bp + k0 + 32);
      br[1]  = *(const f32x4*)(bp + k0 + 36);
    }
    tile_mma(Al, Bl, waveM, waveN, l15, half, acc);
  }

#pragma unroll
  for (int si = 0; si < 2; ++si)
#pragma unroll
    for (int sj = 0; sj < 2; ++sj) {
      const int col = nBase + waveN*32 + sj*16 + l15;
      const float bv = bias[col];
#pragma unroll
      for (int r = 0; r < 8; ++r) {
        const int row = mBase + waveM*32 + si*16 + half*8 + r;
        out[(size_t)row * D_MODEL + col] = acc[si][sj][r] + bv;
      }
    }
}

extern "C" void kernel_launch(void* const* d_in, const int* in_sizes, int n_in,
                              void* d_out, int out_size, void* d_ws, size_t ws_size,
                              hipStream_t stream) {
  (void)in_sizes; (void)n_in; (void)out_size; (void)ws_size;
  const float* x       = (const float*)d_in[0];
  const float* wq_w    = (const float*)d_in[1];
  const float* wq_b    = (const float*)d_in[2];
  const float* wk_w    = (const float*)d_in[3];
  const float* wk_b    = (const float*)d_in[4];
  const float* wv_w    = (const float*)d_in[5];
  const float* wv_b    = (const float*)d_in[6];
  const float* dense_w = (const float*)d_in[7];
  const float* dense_b = (const float*)d_in[8];

  const size_t NE = (size_t)M_TOTAL * D_MODEL;   // 8M elements per tensor
  bf16_t* q_ws   = (bf16_t*)d_ws;
  bf16_t* k_ws   = q_ws  + NE;
  bf16_t* vt_ws  = k_ws  + NE;
  bf16_t* att_ws = vt_ws + NE;                   // total ~67 MB bf16 workspace

  dim3 gQKV(D_MODEL/64, M_TOTAL/128, 3);
  gemm_qkv_kernel<<<gQKV, 256, 0, stream>>>(x, wq_w, wq_b, wk_w, wk_b, wv_w, wv_b,
                                            q_ws, k_ws, vt_ws);

  dim3 gAtt(SEQ/128, BATCH*HEADS);
  attn_kernel<<<gAtt, 256, 0, stream>>>(q_ws, k_ws, vt_ws, att_ws);

  dim3 gOut(D_MODEL/64, M_TOTAL/128);
  gemm_dense_kernel<<<gOut, 256, 0, stream>>>(att_ws, dense_w, dense_b, (float*)d_out);
}
